// Mlp_57002805952962
// MI455X (gfx1250) — matrix operand order repaired and checked by reading, not compiled, not run on hardware
//
#include <hip/hip_runtime.h>

#define HID   1024
#define FDIM  512
#define NEXP  32
#define NTOK  2048
#define CAP   2048

typedef __attribute__((ext_vector_type(16))) _Float16 v16h;
typedef __attribute__((ext_vector_type(8)))  float    v8f;
typedef int v4i_vs __attribute__((vector_size(16)));   // pointee type of async-LDS builtin

union Frag16 { v16h v; uint4 u[2]; };

#if __has_builtin(__builtin_amdgcn_global_load_async_to_lds_b128) && \
    __has_builtin(__builtin_amdgcn_s_wait_asynccnt)
#define ASYNC_LDS 1
#endif

// 16-byte global -> LDS copy. Async (ASYNCcnt-tracked) on CDNA5 when available.
__device__ __forceinline__ void cp16(const void* g, void* l) {
#ifdef ASYNC_LDS
    __builtin_amdgcn_global_load_async_to_lds_b128(
        (__attribute__((address_space(1))) v4i_vs*)(uintptr_t)g,
        (__attribute__((address_space(3))) v4i_vs*)(unsigned)(uintptr_t)l, 0, 0);
#else
    *(uint4*)l = *(const uint4*)g;
#endif
}
// Wait until at most N (if nxt) / 0 async copies outstanding for this wave.
template <int N>
__device__ __forceinline__ void cpwaitT(bool nxt) {
#ifdef ASYNC_LDS
    if (nxt) __builtin_amdgcn_s_wait_asynccnt(N);
    else     __builtin_amdgcn_s_wait_asynccnt(0);
#endif
}

// A-matrix 16x32 f16 frag (lane&15 = M; lane bit4: K {0..7,16..23} vs {8..15,24..31})
__device__ __forceinline__ v16h load_frag_a(const _Float16* rowbase, int lane) {
    int c0 = (lane & 16) ? 8 : 0;
    Frag16 f;
    f.u[0] = *(const uint4*)(rowbase + c0);
    f.u[1] = *(const uint4*)(rowbase + c0 + 16);
    return f.v;
}
// B-matrix 32x16 f16 frag (lane&15 = N; lane bit4: K {0..7,16..23} vs {8..15,24..31})
__device__ __forceinline__ v16h load_frag_b(const _Float16* rowbase, int lane) {
    int c0 = (lane & 16) ? 8 : 0;
    Frag16 f;
    f.u[0] = *(const uint4*)(rowbase + c0);
    f.u[1] = *(const uint4*)(rowbase + c0 + 16);
    return f.v;
}

__device__ __forceinline__ float silu_mul(float g, float u) {
    return u * g * __builtin_amdgcn_rcpf(1.f + __expf(-g));
}

// ---------------- kernel 0: zero output + expert counters ----------------
__global__ void moe_init_kernel(float* __restrict__ out, int* __restrict__ counts) {
    int i = blockIdx.x * 256 + threadIdx.x;
    ((float4*)out)[i] = make_float4(0.f, 0.f, 0.f, 0.f);
    if (i < NEXP) counts[i] = 0;
}

// ---------------- fp32 -> fp16 bulk convert (8 elem / thread) -----------
__global__ void moe_cvt_kernel(const float* __restrict__ src, _Float16* __restrict__ dst) {
    size_t b = ((size_t)blockIdx.x * 256 + threadIdx.x) * 8;
    float4 a = *(const float4*)(src + b);
    float4 c = *(const float4*)(src + b + 4);
    union { _Float16 h[8]; uint4 u; } t;
    t.h[0] = (_Float16)a.x; t.h[1] = (_Float16)a.y; t.h[2] = (_Float16)a.z; t.h[3] = (_Float16)a.w;
    t.h[4] = (_Float16)c.x; t.h[5] = (_Float16)c.y; t.h[6] = (_Float16)c.z; t.h[7] = (_Float16)c.w;
    *(uint4*)(dst + b) = t.u;
}

// ---------------- kernel 1: router (softmax + top-8 + compaction) --------
__global__ void moe_router_kernel(const float* __restrict__ x,
                                  const float* __restrict__ gw,
                                  int* __restrict__ counts,
                                  int* __restrict__ etok,
                                  float* __restrict__ ewt) {
    int t = blockIdx.x;
    int lane = threadIdx.x;                            // 32 threads, lane == expert id
    __shared__ float xs[HID];
    const float* xrow = x + (size_t)t * HID;
    for (int i = lane; i < HID / 4; i += 32)
        ((float4*)xs)[i] = ((const float4*)xrow)[i];
    __syncthreads();

    const float* wrow = gw + (size_t)lane * HID;
    float acc = 0.f;
    for (int k = 0; k < HID; k += 4) {
        float4 w4 = *(const float4*)(wrow + k);
        acc += xs[k] * w4.x + xs[k + 1] * w4.y + xs[k + 2] * w4.z + xs[k + 3] * w4.w;
    }
    float mx = acc;
    for (int m = 16; m; m >>= 1) mx = fmaxf(mx, __shfl_xor(mx, m, 32));
    float p = __expf(acc - mx);
    float s = p;
    for (int m = 16; m; m >>= 1) s += __shfl_xor(s, m, 32);
    p /= s;
    bool sel = false;
    float wsum = 0.f;
    for (int it = 0; it < 8; ++it) {
        float v = sel ? -1.f : p;
        int idx = lane;
        for (int m = 16; m; m >>= 1) {
            float ov = __shfl_xor(v, m, 32);
            int   oi = __shfl_xor(idx, m, 32);
            if (ov > v || (ov == v && oi < idx)) { v = ov; idx = oi; }
        }
        if (lane == idx) sel = true;
        wsum += v;
    }
    if (sel) {
        int pos = atomicAdd(&counts[lane], 1);
        etok[lane * CAP + pos] = t;
        ewt[lane * CAP + pos]  = p / wsum;
    }
}

// ---------------- kernel 2: gate_up GEMM + SiLU ---------------------------
// 256 thr / 8 waves; tile: 64 tokens x 128 h-cols (128 g + 128 u outputs); K=1024
// wave (wm 0..3, wn 0..1): rows wm*16, 4 g-tiles + 4 u-tiles of 16 cols.
template <bool W16>
__global__ __launch_bounds__(256) void moe_gateup_kernel(
        const _Float16* __restrict__ xh, const float* __restrict__ guw,
        const _Float16* __restrict__ guw16,
        const int* __restrict__ counts, const int* __restrict__ etok,
        _Float16* __restrict__ hbuf) {
    int e   = blockIdx.x >> 5;
    int mt  = blockIdx.x & 31;
    int cnt = counts[e];
    int m0  = mt * 64;
    if (m0 >= cnt) return;
    int f0 = blockIdx.y * 128;

    // LDS pool: A bufs 2x5120B | B bufs 2x20480B; Hs (17408B) aliases B after loop
    __shared__ __attribute__((aligned(16))) char smem[2 * 5120 + 2 * 20480];
    __shared__ int tokS[64];
    _Float16* AsB = (_Float16*)smem;             // + buf*2560 halves
    _Float16* BsB = (_Float16*)(smem + 10240);   // + buf*10240 halves
    _Float16* Hs  = (_Float16*)(smem + 10240);

    int tid = threadIdx.x, lane = tid & 31, w = tid >> 5;
    int wm = w >> 1, wn = w & 1;

    if (tid < 64) {
        int idx = m0 + tid;
        tokS[tid] = (idx < cnt) ? etok[e * CAP + idx] : 0;  // clamp: junk rows discarded later
    }
    __syncthreads();

    int arow = tid >> 2, ach = (tid & 3) * 8;      // A: 64 rows x 4 chunks of 8 halves
    const _Float16* xrow = xh + (size_t)tokS[arow] * HID + ach;
    int aofs = arow * 40 + ach;

    int brow = tid >> 3, bc4 = (tid & 7) * 4;      // B f32 path: 32 rows/pass x 8 float4
    const float* bptr[8];
    const _Float16* bsrc[4];
    int bofs[4];
    float4 pb[8];
    if (W16) {
#pragma unroll
        for (int q = 0; q < 4; ++q) {
            int chunk = q * 256 + tid;
            int lr = chunk >> 2, ch = chunk & 3;
            int gr = (lr < 128) ? (f0 + lr) : (FDIM + f0 + (lr - 128));
            bsrc[q] = guw16 + ((size_t)e * 1024 + gr) * HID + ch * 8;
            bofs[q] = lr * 40 + ch * 8;
        }
#pragma unroll
        for (int q = 0; q < 4; ++q) cp16(bsrc[q], BsB + bofs[q]);
    } else {
#pragma unroll
        for (int p = 0; p < 8; ++p) {
            int lr = p * 32 + brow;
            int gr = (lr < 128) ? (f0 + lr) : (FDIM + f0 + (lr - 128));
            bptr[p] = guw + ((size_t)e * 1024 + gr) * HID + bc4;
            pb[p] = *(const float4*)bptr[p];
        }
    }
    cp16(xrow, AsB + aofs);                        // A prefetch for kk=0

    v8f ag[4] = {}, au[4] = {};

    for (int i = 0; i < HID / 32; ++i) {
        int kk = i * 32, buf = i & 1;
        bool nxt = (i + 1) < HID / 32;
        _Float16* Ac = AsB + buf * 2560;
        _Float16* Bc = BsB + buf * 10240;
        _Float16* An = AsB + (buf ^ 1) * 2560;
        _Float16* Bn = BsB + (buf ^ 1) * 10240;
        if (!W16) {                                 // convert-store this step's B
#pragma unroll
            for (int p = 0; p < 8; ++p) {
                _Float16* bp = Bc + (p * 32 + brow) * 40 + bc4;
                bp[0] = (_Float16)pb[p].x; bp[1] = (_Float16)pb[p].y;
                bp[2] = (_Float16)pb[p].z; bp[3] = (_Float16)pb[p].w;
            }
        }
        if (nxt) {                                  // prefetch next step into other buffer
            cp16(xrow + kk + 32, An + aofs);
            if (W16) {
#pragma unroll
                for (int q = 0; q < 4; ++q) cp16(bsrc[q] + kk + 32, Bn + bofs[q]);
            } else {
#pragma unroll
                for (int p = 0; p < 8; ++p) pb[p] = *(const float4*)(bptr[p] + kk + 32);
            }
        }
        cpwaitT<W16 ? 5 : 1>(nxt);
        __syncthreads();

        v16h a = load_frag_a(Ac + (wm * 16 + (lane & 15)) * 40, lane);
#pragma unroll
        for (int j = 0; j < 4; ++j) {
            v16h bg = load_frag_b(Bc + (wn * 64 + j * 16 + (lane & 15)) * 40, lane);
            ag[j] = __builtin_amdgcn_wmma_f32_16x16x32_f16(false, a, false, bg, (short)0, ag[j], false, false);
        }
#pragma unroll
        for (int j = 0; j < 4; ++j) {
            v16h bu = load_frag_b(Bc + (128 + wn * 64 + j * 16 + (lane & 15)) * 40, lane);
            au[j] = __builtin_amdgcn_wmma_f32_16x16x32_f16(false, a, false, bu, (short)0, au[j], false, false);
        }
        __syncthreads();
    }

    __syncthreads();                                // B bufs dead -> reuse as Hs
    int ml = (lane >> 4) << 3, nl = lane & 15;
#pragma unroll
    for (int j = 0; j < 4; ++j) {
#pragma unroll
        for (int r = 0; r < 8; ++r) {
            int m = wm * 16 + r + ml;
            Hs[m * 136 + wn * 64 + j * 16 + nl] = (_Float16)silu_mul(ag[j][r], au[j][r]);
        }
    }
    __syncthreads();
    int row = tid >> 2, c0 = (tid & 3) * 32;        // 64 rows x 128 halves writeback
    const uint4* src = (const uint4*)&Hs[row * 136 + c0];
    uint4* dst = (uint4*)&hbuf[((size_t)e * CAP + m0 + row) * FDIM + f0 + c0];
    dst[0] = src[0]; dst[1] = src[1]; dst[2] = src[2]; dst[3] = src[3];
}

// ---------------- kernel 3: down GEMM + weighted scatter-add --------------
// 256 thr / 8 waves; tile: 64 slots x 128 out-cols; K=512
template <bool W16>
__global__ __launch_bounds__(256) void moe_down_kernel(
        const _Float16* __restrict__ hbuf, const float* __restrict__ dw,
        const _Float16* __restrict__ dw16,
        const int* __restrict__ counts, const int* __restrict__ etok,
        const float* __restrict__ ewt, float* __restrict__ out) {
    int e   = blockIdx.x >> 5;
    int mt  = blockIdx.x & 31;
    int cnt = counts[e];
    int m0  = mt * 64;
    if (m0 >= cnt) return;
    int n0 = blockIdx.y * 128;

    // LDS pool: A bufs 2x5120B | B bufs 2x10240B = 30720B; Ys (64*132*4B) aliases all
    __shared__ __attribute__((aligned(16))) char smem[64 * 132 * 4];
    __shared__ int   tokS[64];
    __shared__ float wtS[64];
    _Float16* AsB = (_Float16*)smem;             // + buf*2560 halves
    _Float16* BsB = (_Float16*)(smem + 10240);   // + buf*5120 halves
    float* Ys = (float*)smem;

    int tid = threadIdx.x, lane = tid & 31, w = tid >> 5;
    int wm = w >> 1, wn = w & 1;

    if (tid < 64) {
        int idx = m0 + tid;
        bool v = idx < cnt;
        tokS[tid] = v ? etok[e * CAP + idx] : -1;
        wtS[tid]  = v ? ewt[e * CAP + idx] : 0.f;
    }
    __syncthreads();

    int arow = tid >> 2, ach = (tid & 3) * 8;
    const _Float16* hrow = hbuf + ((size_t)e * CAP + m0 + arow) * FDIM + ach;
    int aofs = arow * 40 + ach;

    int brow = tid >> 3, bc4 = (tid & 7) * 4;
    const float* bptr[4];
    const _Float16* bsrc[2];
    int bofs[2];
    float4 pb[4];
    if (W16) {
#pragma unroll
        for (int q = 0; q < 2; ++q) {
            int chunk = q * 256 + tid;
            int lr = chunk >> 2, ch = chunk & 3;
            bsrc[q] = dw16 + ((size_t)e * HID + n0 + lr) * FDIM + ch * 8;
            bofs[q] = lr * 40 + ch * 8;
        }
#pragma unroll
        for (int q = 0; q < 2; ++q) cp16(bsrc[q], BsB + bofs[q]);
    } else {
#pragma unroll
        for (int p = 0; p < 4; ++p) {
            bptr[p] = dw + ((size_t)e * HID + n0 + p * 32 + brow) * FDIM + bc4;
            pb[p] = *(const float4*)bptr[p];
        }
    }
    cp16(hrow, AsB + aofs);

    v8f acc[4] = {};

    for (int i = 0; i < FDIM / 32; ++i) {
        int kk = i * 32, buf = i & 1;
        bool nxt = (i + 1) < FDIM / 32;
        _Float16* Ac = AsB + buf * 2560;
        _Float16* Bc = BsB + buf * 5120;
        _Float16* An = AsB + (buf ^ 1) * 2560;
        _Float16* Bn = BsB + (buf ^ 1) * 5120;
        if (!W16) {
#pragma unroll
            for (int p = 0; p < 4; ++p) {
                _Float16* bp = Bc + (p * 32 + brow) * 40 + bc4;
                bp[0] = (_Float16)pb[p].x; bp[1] = (_Float16)pb[p].y;
                bp[2] = (_Float16)pb[p].z; bp[3] = (_Float16)pb[p].w;
            }
        }
        if (nxt) {
            cp16(hrow + kk + 32, An + aofs);
            if (W16) {
#pragma unroll
                for (int q = 0; q < 2; ++q) cp16(bsrc[q] + kk + 32, Bn + bofs[q]);
            } else {
#pragma unroll
                for (int p = 0; p < 4; ++p) pb[p] = *(const float4*)(bptr[p] + kk + 32);
            }
        }
        cpwaitT<W16 ? 3 : 1>(nxt);
        __syncthreads();

        v16h a = load_frag_a(Ac + (wm * 16 + (lane & 15)) * 40, lane);
#pragma unroll
        for (int j = 0; j < 4; ++j) {
            v16h b = load_frag_b(Bc + (wn * 64 + j * 16 + (lane & 15)) * 40, lane);
            acc[j] = __builtin_amdgcn_wmma_f32_16x16x32_f16(false, a, false, b, (short)0, acc[j], false, false);
        }
        __syncthreads();
    }

    __syncthreads();                                // buffers dead -> reuse as Ys
    int ml = (lane >> 4) << 3, nl = lane & 15;
#pragma unroll
    for (int j = 0; j < 4; ++j) {
#pragma unroll
        for (int r = 0; r < 8; ++r) {
            int m = wm * 16 + r + ml;
            Ys[m * 132 + wn * 64 + j * 16 + nl] = acc[j][r];
        }
    }
    __syncthreads();
    int row = tid >> 2, c0 = (tid & 3) * 32;
    int tok = tokS[row];
    float wt = wtS[row];
    if (tok >= 0) {
        float* op = out + (size_t)tok * HID + n0 + c0;
#pragma unroll
        for (int i = 0; i < 32; ++i)
            atomicAdd(op + i, wt * Ys[row * 132 + c0 + i]);
    }
}

// ---------------- launcher ----------------
#define WS_COUNTS 0
#define WS_ETOK   ((size_t)256)
#define WS_EWT    (WS_ETOK + (size_t)NEXP * CAP * 4)
#define WS_XH     (WS_EWT + (size_t)NEXP * CAP * 4)                  // 524544
#define WS_HBUF   (WS_XH + (size_t)NTOK * HID * 2)                   // + 4 MB
#define WS_GW16   (WS_HBUF + (size_t)NEXP * CAP * FDIM * 2)          // + 64 MB
#define WS_DW16   (WS_GW16 + (size_t)NEXP * 2 * FDIM * HID * 2)      // + 64 MB
#define WS_FULL   (WS_DW16 + (size_t)NEXP * HID * FDIM * 2)          // + 32 MB

extern "C" void kernel_launch(void* const* d_in, const int* in_sizes, int n_in,
                              void* d_out, int out_size, void* d_ws, size_t ws_size,
                              hipStream_t stream) {
    const float* x   = (const float*)d_in[0];   // [1,2048,1024]
    const float* gw  = (const float*)d_in[1];   // [32,1024]
    const float* guw = (const float*)d_in[2];   // [32,1024,1024]
    const float* dw  = (const float*)d_in[3];   // [32,1024,512]
    float* out = (float*)d_out;

    char* ws = (char*)d_ws;
    int*      counts = (int*)(ws + WS_COUNTS);
    int*      etok   = (int*)(ws + WS_ETOK);
    float*    ewt    = (float*)(ws + WS_EWT);
    _Float16* xh     = (_Float16*)(ws + WS_XH);
    _Float16* hbuf   = (_Float16*)(ws + WS_HBUF);
    _Float16* gw16   = (_Float16*)(ws + WS_GW16);
    _Float16* dw16   = (_Float16*)(ws + WS_DW16);

    bool w16 = ws_size >= WS_FULL;                 // deterministic path choice

    moe_init_kernel<<<(NTOK * HID / 4) / 256, 256, 0, stream>>>(out, counts);
    moe_cvt_kernel<<<(NTOK * HID) / (8 * 256), 256, 0, stream>>>(x, xh);
    if (w16) {
        moe_cvt_kernel<<<(NEXP * 2 * FDIM * HID) / (8 * 256), 256, 0, stream>>>(guw, gw16);
        moe_cvt_kernel<<<(NEXP * HID * FDIM) / (8 * 256), 256, 0, stream>>>(dw, dw16);
    }
    moe_router_kernel<<<NTOK, 32, 0, stream>>>(x, gw, counts, etok, ewt);
    if (w16) {
        moe_gateup_kernel<true><<<dim3(NEXP * 32, FDIM / 128), 256, 0, stream>>>(
            xh, guw, gw16, counts, etok, hbuf);
        moe_down_kernel<true><<<dim3(NEXP * 32, HID / 128), 256, 0, stream>>>(
            hbuf, dw, dw16, counts, etok, ewt, out);
    } else {
        moe_gateup_kernel<false><<<dim3(NEXP * 32, FDIM / 128), 256, 0, stream>>>(
            xh, guw, gw16, counts, etok, hbuf);
        moe_down_kernel<false><<<dim3(NEXP * 32, HID / 128), 256, 0, stream>>>(
            hbuf, dw, dw16, counts, etok, ewt, out);
    }
}
